// SE3EquivariantGNN_65584150610130
// MI455X (gfx1250) — matrix operand order repaired and checked by reading, not compiled, not run on hardware
//
#include <hip/hip_runtime.h>

typedef __attribute__((ext_vector_type(16))) _Float16 h16;
typedef __attribute__((ext_vector_type(8)))  _Float16 h8;
typedef __attribute__((ext_vector_type(8)))  float    fx8;

#define DEVI static __device__ __forceinline__

// ---- WMMA fragment helpers (gfx1250 wave32 layouts, CDNA5 ISA 7.12.2) ----
// A 16x32 f16: lane m = lane%16 ; elements j<8 -> K = 8*half+j ; j>=8 -> K = 16+8*half+(j-8)
DEVI h16 afrag(const _Float16* __restrict__ arow, int k0, int hf) {
  union { h16 v; h8 h[2]; } u;
  u.h[0] = *(const h8*)(arow + k0 + 8*hf);
  u.h[1] = *(const h8*)(arow + k0 + 16 + 8*hf);
  return u.v;
}
// B 32x16 f16: lane n = lane%16 ; elements j<8 -> K = 8*half+j ; j>=8 -> K = 16+8*half+(j-8)
DEVI h16 bfrag(const _Float16* __restrict__ B, int ldb, int k, int n0) {
  const int ln = threadIdx.x & 31;
  const _Float16* q = B + (size_t)(k - ln + 8 * (ln >> 4)) * ldb + n0 + (ln & 15);
  h16 v;
  #pragma unroll
  for (int j = 0; j < 8; ++j) {
    v[j] = q[(size_t)j * ldb];
    v[8 + j] = q[(size_t)(16 + j) * ldb];
  }
  return v;
}
DEVI fx8 wmma16(h16 a, h16 b, fx8 c) {
  return __builtin_amdgcn_wmma_f32_16x16x32_f16(false, a, false, b, (short)0, c, false, false);
}
DEVI float sigm(float x) { return 1.f / (1.f + __expf(-x)); }

// =====================================================================
// Generic GEMM: Out[M,NC] = act(A[M,128] @ W[128,NC] + bias), W staged in LDS
// ACT: 0=none 1=silu 2=relu
// =====================================================================
template<int NT, int ACT, bool OUTF16>
__global__ __launch_bounds__(256) void k_gemm(
    const _Float16* __restrict__ A, const _Float16* __restrict__ W,
    const float* __restrict__ bias, void* __restrict__ Out, int ldo, int Mrows)
{
  constexpr int K = 128, NC = NT * 16, PITCH = NC + 8;
  __shared__ _Float16 sW[K * PITCH];
  const int tid = threadIdx.x;
  for (int i = tid; i < K * NC; i += 256) {
    int k = i / NC, n = i - k * NC;
    sW[k * PITCH + n] = W[i];
  }
  __syncthreads();
  const int lane = tid & 31, wid = tid >> 5;
  const int hf = lane >> 4, nl = lane & 15;
  const int ntiles = Mrows >> 4;
  for (int t = blockIdx.x * 8 + wid; t < ntiles; t += gridDim.x * 8) {
    const _Float16* ar = A + (size_t)(t * 16 + nl) * K;
    int tn = t + gridDim.x * 8;
    if (tn < ntiles) __builtin_prefetch(A + (size_t)(tn * 16 + nl) * K, 0, 1);
    fx8 acc[NT];
    #pragma unroll
    for (int ct = 0; ct < NT; ++ct) {
      float bv = bias[ct * 16 + nl];
      #pragma unroll
      for (int r = 0; r < 8; ++r) acc[ct][r] = bv;
    }
    #pragma unroll
    for (int k0 = 0; k0 < K; k0 += 32) {
      h16 af = afrag(ar, k0, hf);
      #pragma unroll
      for (int ct = 0; ct < NT; ++ct)
        acc[ct] = wmma16(af, bfrag(sW, PITCH, k0 + lane, ct * 16), acc[ct]);
    }
    #pragma unroll
    for (int r = 0; r < 8; ++r) {
      int row = t * 16 + 8 * hf + r;
      #pragma unroll
      for (int ct = 0; ct < NT; ++ct) {
        int n = nl + 16 * ct;
        float xv = acc[ct][r];
        if (ACT == 1) xv = xv * sigm(xv);
        if (ACT == 2) xv = fmaxf(xv, 0.f);
        if constexpr (OUTF16) ((_Float16*)Out)[(size_t)row * ldo + n] = (_Float16)xv;
        else                  ((float*)Out)[(size_t)row * ldo + n] = xv;
      }
    }
  }
}

// =====================================================================
// e_s = h1 @ ew2 + eb2 ; also p = e_s * s16[src]  (fused gather epilogue)
// =====================================================================
__global__ __launch_bounds__(256) void k_es_p(
    const _Float16* __restrict__ A, const _Float16* __restrict__ W,
    const float* __restrict__ bias, _Float16* __restrict__ es,
    _Float16* __restrict__ p, const _Float16* __restrict__ s16,
    const int* __restrict__ src, int Mrows)
{
  constexpr int K = 128, NC = 128, PITCH = 136;
  __shared__ _Float16 sW[K * PITCH];
  const int tid = threadIdx.x;
  for (int i = tid; i < K * NC; i += 256) sW[(i >> 7) * PITCH + (i & 127)] = W[i];
  __syncthreads();
  const int lane = tid & 31, wid = tid >> 5;
  const int hf = lane >> 4, nl = lane & 15;
  const int ntiles = Mrows >> 4;
  for (int t = blockIdx.x * 8 + wid; t < ntiles; t += gridDim.x * 8) {
    const _Float16* ar = A + (size_t)(t * 16 + nl) * K;
    fx8 acc[8];
    #pragma unroll
    for (int ct = 0; ct < 8; ++ct) {
      float bv = bias[ct * 16 + nl];
      #pragma unroll
      for (int r = 0; r < 8; ++r) acc[ct][r] = bv;
    }
    #pragma unroll
    for (int k0 = 0; k0 < K; k0 += 32) {
      h16 af = afrag(ar, k0, hf);
      #pragma unroll
      for (int ct = 0; ct < 8; ++ct)
        acc[ct] = wmma16(af, bfrag(sW, PITCH, k0 + lane, ct * 16), acc[ct]);
    }
    #pragma unroll
    for (int r = 0; r < 8; ++r) {
      int row = t * 16 + 8 * hf + r;
      size_t sb = (size_t)src[row] * 128;
      #pragma unroll
      for (int ct = 0; ct < 8; ++ct) {
        int n = nl + 16 * ct;
        float xv = acc[ct][r];
        es[(size_t)row * 128 + n] = (_Float16)xv;
        p [(size_t)row * 128 + n] = (_Float16)(xv * (float)s16[sb + n]);
      }
    }
  }
}

// =====================================================================
// m0 = p @ U1 (K=128)  +  dots @ U2p (K=64, zero-padded) ; atomic scatter to a0[dst]
// =====================================================================
__global__ __launch_bounds__(256) void k_m0(
    const _Float16* __restrict__ P, const _Float16* __restrict__ D,
    const _Float16* __restrict__ U1, const _Float16* __restrict__ U2,
    const int* __restrict__ dst, float* __restrict__ a0, int Mrows)
{
  constexpr int PITCH = 136;
  __shared__ _Float16 sW[(128 + 64) * PITCH];
  const int tid = threadIdx.x;
  for (int i = tid; i < 128 * 128; i += 256) sW[(i >> 7) * PITCH + (i & 127)] = U1[i];
  for (int i = tid; i < 64 * 128;  i += 256) sW[(128 + (i >> 7)) * PITCH + (i & 127)] = U2[i];
  __syncthreads();
  const int lane = tid & 31, wid = tid >> 5;
  const int hf = lane >> 4, nl = lane & 15;
  const int ntiles = Mrows >> 4;
  for (int t = blockIdx.x * 8 + wid; t < ntiles; t += gridDim.x * 8) {
    const _Float16* pr = P + (size_t)(t * 16 + nl) * 128;
    const _Float16* dr = D + (size_t)(t * 16 + nl) * 64;
    fx8 acc[8];
    #pragma unroll
    for (int ct = 0; ct < 8; ++ct)
      #pragma unroll
      for (int r = 0; r < 8; ++r) acc[ct][r] = 0.f;
    #pragma unroll
    for (int k0 = 0; k0 < 128; k0 += 32) {
      h16 af = afrag(pr, k0, hf);
      #pragma unroll
      for (int ct = 0; ct < 8; ++ct)
        acc[ct] = wmma16(af, bfrag(sW, PITCH, k0 + lane, ct * 16), acc[ct]);
    }
    #pragma unroll
    for (int k0 = 0; k0 < 64; k0 += 32) {
      h16 af = afrag(dr, k0, hf);
      #pragma unroll
      for (int ct = 0; ct < 8; ++ct)
        acc[ct] = wmma16(af, bfrag(sW, PITCH, 128 + k0 + lane, ct * 16), acc[ct]);
    }
    #pragma unroll
    for (int r = 0; r < 8; ++r) {
      int row = t * 16 + 8 * hf + r;
      size_t ob = (size_t)dst[row] * 128;
      #pragma unroll
      for (int ct = 0; ct < 8; ++ct)
        atomicAdd(&a0[ob + nl + 16 * ct], acc[ct][r]);
    }
  }
}

// =====================================================================
// a = s16[src] @ V1p ; b = es @ V2p (both [E,48], K=128)
// m1 = a*rel + vs*b + (vs x rel)*wc -> atomic scatter into a1[dst]
// =====================================================================
__global__ __launch_bounds__(256) void k_ab_m1(
    const _Float16* __restrict__ s16, const _Float16* __restrict__ es,
    const _Float16* __restrict__ V1, const _Float16* __restrict__ V2,
    const float* __restrict__ wc, const int* __restrict__ src,
    const int* __restrict__ dst, const float* __restrict__ rel,
    const float* __restrict__ v, float* __restrict__ a1, int Mrows)
{
  constexpr int PITCH = 56;
  __shared__ _Float16 sV[2 * 128 * PITCH];
  __shared__ float swc[48];
  const int tid = threadIdx.x;
  for (int i = tid; i < 128 * 48; i += 256) {
    int k = i / 48, n = i - k * 48;
    sV[k * PITCH + n] = V1[i];
    sV[(128 + k) * PITCH + n] = V2[i];
  }
  if (tid < 48) swc[tid] = (tid < 42) ? wc[tid] : 0.f;
  __syncthreads();
  const int lane = tid & 31, wid = tid >> 5;
  const int hf = lane >> 4, nl = lane & 15;
  const int ntiles = Mrows >> 4;
  for (int t = blockIdx.x * 8 + wid; t < ntiles; t += gridDim.x * 8) {
    const _Float16* arS = s16 + (size_t)src[t * 16 + nl] * 128;
    const _Float16* arE = es + (size_t)(t * 16 + nl) * 128;
    fx8 accA[3], accB[3];
    #pragma unroll
    for (int ct = 0; ct < 3; ++ct)
      #pragma unroll
      for (int r = 0; r < 8; ++r) { accA[ct][r] = 0.f; accB[ct][r] = 0.f; }
    #pragma unroll
    for (int k0 = 0; k0 < 128; k0 += 32) {
      h16 afS = afrag(arS, k0, hf);
      h16 afE = afrag(arE, k0, hf);
      #pragma unroll
      for (int ct = 0; ct < 3; ++ct) {
        accA[ct] = wmma16(afS, bfrag(sV, PITCH, k0 + lane, ct * 16), accA[ct]);
        accB[ct] = wmma16(afE, bfrag(sV, PITCH, 128 + k0 + lane, ct * 16), accB[ct]);
      }
    }
    #pragma unroll
    for (int r = 0; r < 8; ++r) {
      int row = t * 16 + 8 * hf + r;
      int sr = src[row], dr = dst[row];
      float rx = rel[(size_t)row * 3 + 0];
      float ry = rel[(size_t)row * 3 + 1];
      float rz = rel[(size_t)row * 3 + 2];
      const float* vr = v + (size_t)sr * 126;
      float* ao = a1 + (size_t)dr * 126;
      #pragma unroll
      for (int ct = 0; ct < 3; ++ct) {
        int c = nl + 16 * ct;
        if (c < 42) {
          float av = accA[ct][r], bv = accB[ct][r], w = swc[c];
          float vx = vr[c * 3 + 0], vy = vr[c * 3 + 1], vz = vr[c * 3 + 2];
          float cx = vy * rz - vz * ry;
          float cy = vz * rx - vx * rz;
          float cz = vx * ry - vy * rx;
          atomicAdd(&ao[c * 3 + 0], av * rx + vx * bv + cx * w);
          atomicAdd(&ao[c * 3 + 1], av * ry + vy * bv + cy * w);
          atomicAdd(&ao[c * 3 + 2], av * rz + vz * bv + cz * w);
        }
      }
    }
  }
}

// ------------------- small elementwise / setup kernels -------------------
__global__ void k_zero(float* p, int n) {
  int i = blockIdx.x * 256 + threadIdx.x;
  if (i < n) p[i] = 0.f;
}
__global__ void k_cvt(const float* __restrict__ s, _Float16* __restrict__ d, int n) {
  int i = blockIdx.x * 256 + threadIdx.x;
  if (i < n) d[i] = (_Float16)s[i];
}
__global__ void k_cvt_pad(const float* __restrict__ s, _Float16* __restrict__ d,
                          int rows, int cols, int prows, int pcols) {
  int i = blockIdx.x * 256 + threadIdx.x;
  if (i >= prows * pcols) return;
  int r = i / pcols, c = i - r * pcols;
  d[i] = (r < rows && c < cols) ? (_Float16)s[(size_t)r * cols + c] : (_Float16)0.f;
}
__global__ void k_sinit(const int* __restrict__ x, const float* __restrict__ emb,
                        float* __restrict__ s, _Float16* __restrict__ s16, int N) {
  int i = blockIdx.x * 256 + threadIdx.x;
  if (i >= N * 128) return;
  int n = i >> 7, h = i & 127;
  float val = emb[(size_t)x[n] * 128 + h];
  s[i] = val; s16[i] = (_Float16)val;
}
__global__ void k_eattr(const int* __restrict__ et, const float* __restrict__ emb,
                        _Float16* __restrict__ ea, long n) {
  long i = (long)blockIdx.x * 256 + threadIdx.x;
  if (i >= n) return;
  long e = i >> 7; int h = (int)(i & 127);
  ea[i] = (_Float16)emb[(size_t)et[e] * 128 + h];
}
__global__ void k_rel(const int* __restrict__ src, const int* __restrict__ dst,
                      const float* __restrict__ pos, float* __restrict__ rel, int E) {
  int e = blockIdx.x * 256 + threadIdx.x;
  if (e >= E) return;
  int s = src[e], d = dst[e];
  rel[(size_t)e * 3 + 0] = pos[(size_t)d * 3 + 0] - pos[(size_t)s * 3 + 0];
  rel[(size_t)e * 3 + 1] = pos[(size_t)d * 3 + 1] - pos[(size_t)s * 3 + 1];
  rel[(size_t)e * 3 + 2] = pos[(size_t)d * 3 + 2] - pos[(size_t)s * 3 + 2];
}
__global__ void k_cnt(const int* __restrict__ dst, float* __restrict__ cnt, int E) {
  int e = blockIdx.x * 256 + threadIdx.x;
  if (e < E) atomicAdd(&cnt[dst[e]], 1.f);
}
__global__ void k_dots(const float* __restrict__ v, const float* __restrict__ rel,
                       const int* __restrict__ src, _Float16* __restrict__ dots, int E) {
  long i = (long)blockIdx.x * 256 + threadIdx.x;
  if (i >= (long)E * 64) return;
  long e = i >> 6; int c = (int)(i & 63);
  _Float16 o = (_Float16)0.f;
  if (c < 42) {
    const float* vr = v + (size_t)src[e] * 126 + c * 3;
    const float* rr = rel + (size_t)e * 3;
    o = (_Float16)(vr[0] * rr[0] + vr[1] * rr[1] + vr[2] * rr[2]);
  }
  dots[i] = o;
}
__global__ __launch_bounds__(128) void k_nodeupd(
    float* __restrict__ s, _Float16* __restrict__ s16, float* __restrict__ v,
    const float* __restrict__ a0, const float* __restrict__ a1,
    const float* __restrict__ cnt, const float* __restrict__ gw,
    const float* __restrict__ gb) {
  __shared__ float sA[128];
  int n = blockIdx.x, t = threadIdx.x;
  float c = cnt[n]; c = c < 1.f ? 1.f : c;
  float a = a0[(size_t)n * 128 + t] / c;
  sA[t] = a;
  float ns = s[(size_t)n * 128 + t] + a * sigm(a);
  s[(size_t)n * 128 + t] = ns;
  s16[(size_t)n * 128 + t] = (_Float16)ns;
  __syncthreads();
  if (t < 42) {
    float acc = gb[t];
    for (int k = 0; k < 128; ++k) acc += sA[k] * gw[(size_t)k * 42 + t];
    float gate = sigm(acc);
    size_t b = (size_t)n * 126 + t * 3;
    v[b + 0] += (a1[b + 0] / c) * gate;
    v[b + 1] += (a1[b + 1] / c) * gate;
    v[b + 2] += (a1[b + 2] / c) * gate;
  }
}
__global__ void k_vpack(const float* __restrict__ v, _Float16* __restrict__ vf, int N) {
  int i = blockIdx.x * 256 + threadIdx.x;
  if (i >= N * 128) return;
  int n = i >> 7, j = i & 127;
  vf[i] = (j < 126) ? (_Float16)v[(size_t)n * 126 + j] : (_Float16)0.f;
}
__global__ void k_poshead(const _Float16* __restrict__ p1, const float* __restrict__ pw2,
                          const float* __restrict__ pb2, const float* __restrict__ pos,
                          float* __restrict__ out, int N) {
  int n = blockIdx.x * 256 + threadIdx.x;
  if (n >= N) return;
  float o0 = pb2[0], o1 = pb2[1], o2 = pb2[2];
  for (int k = 0; k < 128; ++k) {
    float x = (float)p1[(size_t)n * 128 + k];
    o0 += x * pw2[k * 3 + 0];
    o1 += x * pw2[k * 3 + 1];
    o2 += x * pw2[k * 3 + 2];
  }
  out[(size_t)n * 3 + 0] = pos[(size_t)n * 3 + 0] + o0;
  out[(size_t)n * 3 + 1] = pos[(size_t)n * 3 + 1] + o1;
  out[(size_t)n * 3 + 2] = pos[(size_t)n * 3 + 2] + o2;
}

static inline int cdiv(long a, long b) { return (int)((a + b - 1) / b); }

extern "C" void kernel_launch(void* const* d_in, const int* in_sizes, int n_in,
                              void* d_out, int out_size, void* d_ws, size_t ws_size,
                              hipStream_t stream) {
  const int N = in_sizes[0];
  const int E = in_sizes[2];
  const int H = 128, HV = 42, L = 3, HH = 128 * 128;

  const int*   x    = (const int*)d_in[0];
  const int*   src  = (const int*)d_in[1];
  const int*   dst  = src + E;
  const int*   et   = (const int*)d_in[2];
  const float* pos  = (const float*)d_in[3];
  const float* nemb = (const float*)d_in[4];
  const float* eemb = (const float*)d_in[5];
  const float* ew1  = (const float*)d_in[6];
  const float* eb1  = (const float*)d_in[7];
  const float* ew2  = (const float*)d_in[8];
  const float* eb2  = (const float*)d_in[9];
  const float* U1   = (const float*)d_in[10];
  const float* U2   = (const float*)d_in[11];
  const float* V1   = (const float*)d_in[12];
  const float* V2   = (const float*)d_in[13];
  const float* wc   = (const float*)d_in[14];
  const float* gw   = (const float*)d_in[15];
  const float* gb   = (const float*)d_in[16];
  const float* tw1  = (const float*)d_in[17];
  const float* tb1  = (const float*)d_in[18];
  const float* tw2  = (const float*)d_in[19];
  const float* tb2  = (const float*)d_in[20];
  const float* pw1  = (const float*)d_in[21];
  const float* pb1  = (const float*)d_in[22];
  const float* pw2  = (const float*)d_in[23];
  const float* pb2  = (const float*)d_in[24];
  float* out = (float*)d_out;

  // ---- workspace carve ----
  char* ws = (char*)d_ws; size_t off = 0;
  auto carve = [&](size_t bytes) -> void* {
    void* p = ws + off; off += (bytes + 255) & ~(size_t)255; return p;
  };
  _Float16* ew1h = (_Float16*)carve((size_t)L * HH * 2);
  _Float16* ew2h = (_Float16*)carve((size_t)L * HH * 2);
  _Float16* U1h  = (_Float16*)carve((size_t)L * HH * 2);
  _Float16* U2h  = (_Float16*)carve((size_t)L * 64 * 128 * 2);
  _Float16* V1h  = (_Float16*)carve((size_t)L * 128 * 48 * 2);
  _Float16* V2h  = (_Float16*)carve((size_t)L * 128 * 48 * 2);
  _Float16* tw1h = (_Float16*)carve((size_t)HH * 2);
  _Float16* tw2h = (_Float16*)carve((size_t)128 * 32 * 2);
  _Float16* pw1h = (_Float16*)carve((size_t)128 * 128 * 2);
  float*    s    = (float*)carve((size_t)N * 128 * 4);
  _Float16* s16  = (_Float16*)carve((size_t)N * 128 * 2);
  float*    vv   = (float*)carve((size_t)N * 126 * 4);
  float*    cnt  = (float*)carve((size_t)N * 4);
  float*    a0   = (float*)carve((size_t)N * 128 * 4);
  float*    a1   = (float*)carve((size_t)N * 126 * 4);
  _Float16* EA   = (_Float16*)carve((size_t)E * 128 * 2);
  _Float16* h1   = (_Float16*)carve((size_t)E * 128 * 2);
  _Float16* es   = (_Float16*)carve((size_t)E * 128 * 2);
  _Float16* pp   = (_Float16*)carve((size_t)E * 128 * 2);
  float*    rel  = (float*)carve((size_t)E * 3 * 4);
  _Float16* t1   = (_Float16*)carve((size_t)N * 128 * 2);
  _Float16* p1   = (_Float16*)carve((size_t)N * 128 * 2);
  _Float16* vfl  = (_Float16*)carve((size_t)N * 128 * 2);
  _Float16* dots = h1;  // reuse: h1 dead after k_es_p, dots dead after k_m0

  // ---- weight conversions (f32 -> f16, with zero padding where needed) ----
  k_cvt<<<cdiv((long)L * HH, 256), 256, 0, stream>>>(ew1, ew1h, L * HH);
  k_cvt<<<cdiv((long)L * HH, 256), 256, 0, stream>>>(ew2, ew2h, L * HH);
  k_cvt<<<cdiv((long)L * HH, 256), 256, 0, stream>>>(U1, U1h, L * HH);
  k_cvt<<<cdiv(HH, 256), 256, 0, stream>>>(tw1, tw1h, HH);
  k_cvt<<<cdiv(128 * 32, 256), 256, 0, stream>>>(tw2, tw2h, 128 * 32);
  k_cvt_pad<<<cdiv(128 * 128, 256), 256, 0, stream>>>(pw1, pw1h, 126, 128, 128, 128);
  for (int l = 0; l < L; ++l) {
    k_cvt_pad<<<cdiv(64 * 128, 256), 256, 0, stream>>>(U2 + (size_t)l * HV * 128,
                                                       U2h + (size_t)l * 64 * 128, HV, 128, 64, 128);
    k_cvt_pad<<<cdiv(128 * 48, 256), 256, 0, stream>>>(V1 + (size_t)l * 128 * HV,
                                                       V1h + (size_t)l * 128 * 48, 128, HV, 128, 48);
    k_cvt_pad<<<cdiv(128 * 48, 256), 256, 0, stream>>>(V2 + (size_t)l * 128 * HV,
                                                       V2h + (size_t)l * 128 * 48, 128, HV, 128, 48);
  }

  // ---- graph setup ----
  k_zero<<<cdiv((long)N * 126, 256), 256, 0, stream>>>(vv, N * 126);
  k_zero<<<cdiv(N, 256), 256, 0, stream>>>(cnt, N);
  k_sinit<<<cdiv((long)N * 128, 256), 256, 0, stream>>>(x, nemb, s, s16, N);
  k_cnt<<<cdiv(E, 256), 256, 0, stream>>>(dst, cnt, E);
  k_eattr<<<cdiv((long)E * 128, 256), 256, 0, stream>>>(et, eemb, EA, (long)E * 128);
  k_rel<<<cdiv(E, 256), 256, 0, stream>>>(src, dst, pos, rel, E);

  const int GB_E = 1024;                 // E/16 = 40000 row tiles, grid-stride, 8 waves/block
  const int GB_N = cdiv(N / 16, 8);      // node-level GEMMs

  // ---- message-passing layers ----
  for (int l = 0; l < L; ++l) {
    k_zero<<<cdiv((long)N * 128, 256), 256, 0, stream>>>(a0, N * 128);
    k_zero<<<cdiv((long)N * 126, 256), 256, 0, stream>>>(a1, N * 126);
    // h1 = silu(edge_attr @ ew1 + eb1)
    k_gemm<8, 1, true><<<GB_E, 256, 0, stream>>>(EA, ew1h + (size_t)l * HH, eb1 + l * H, h1, 128, E);
    // es = h1 @ ew2 + eb2 ; p = es * s[src]
    k_es_p<<<GB_E, 256, 0, stream>>>(h1, ew2h + (size_t)l * HH, eb2 + l * H, es, pp, s16, src, E);
    // dots = (v[src] . rel), padded to 64 channels
    k_dots<<<cdiv((long)E * 64, 256), 256, 0, stream>>>(vv, rel, src, dots, E);
    // a0[dst] += p @ U1 + dots @ U2
    k_m0<<<GB_E, 256, 0, stream>>>(pp, dots, U1h + (size_t)l * HH, U2h + (size_t)l * 64 * 128,
                                   dst, a0, E);
    // a1[dst] += a*rel + v[src]*b + (v[src] x rel)*wc
    k_ab_m1<<<GB_E, 256, 0, stream>>>(s16, es, V1h + (size_t)l * 128 * 48, V2h + (size_t)l * 128 * 48,
                                      wc + l * HV, src, dst, rel, vv, a1, E);
    // node update: residual + equivariant gate
    k_nodeupd<<<N, 128, 0, stream>>>(s, s16, vv, a0, a1, cnt, gw + (size_t)l * 128 * HV, gb + l * HV);
  }

  // ---- output heads ----
  k_gemm<8, 2, true><<<GB_N, 256, 0, stream>>>(s16, tw1h, tb1, t1, 128, N);
  k_gemm<2, 0, false><<<GB_N, 256, 0, stream>>>(t1, tw2h, tb2, out + (size_t)N * 3, 32, N);
  k_vpack<<<cdiv((long)N * 128, 256), 256, 0, stream>>>(vv, vfl, N);
  k_gemm<8, 2, true><<<GB_N, 256, 0, stream>>>(vfl, pw1h, pb1, p1, 128, N);
  k_poshead<<<cdiv(N, 256), 256, 0, stream>>>(p1, pw2, pb2, pos, out, N);

  (void)n_in; (void)out_size; (void)ws_size;
}
